// TrainedAttention_11501922419155
// MI455X (gfx1250) — matrix operand order repaired and checked by reading, not compiled, not run on hardware
//
#include <hip/hip_runtime.h>

// Problem constants (match reference)
constexpr int Bc  = 4;
constexpr int Tc  = 2048;
constexpr int Hc  = 16;
constexpr int Dvc = 64;
constexpr int DMc = 1024;   // d_model == d_k == H*Dv

// Flip to 0 if the async-to-LDS inline asm mnemonic is rejected.
#define GEMM_USE_ASYNC 1

typedef __attribute__((ext_vector_type(16))) __bf16 bf16x16;
typedef __attribute__((ext_vector_type(8)))  __bf16 bf16x8;
typedef __attribute__((ext_vector_type(8)))  float  f32x8;

__device__ inline f32x8 zero8() {
  f32x8 z = {0.f, 0.f, 0.f, 0.f, 0.f, 0.f, 0.f, 0.f};
  return z;
}

__device__ inline __bf16 f2bf(float f) {
  unsigned u = __float_as_uint(f);
  u += 0x7FFFu + ((u >> 16) & 1u);          // round to nearest even
  unsigned short s = (unsigned short)(u >> 16);
  __bf16 r;
  __builtin_memcpy(&r, &s, sizeof(r));
  return r;
}

__device__ inline bf16x8 ld8(const __bf16* p) {
  return *reinterpret_cast<const bf16x8*>(p);
}

__device__ inline bf16x16 cat16(bf16x8 a, bf16x8 b) {
  return __builtin_shufflevector(a, b, 0, 1, 2, 3, 4, 5, 6, 7,
                                        8, 9, 10, 11, 12, 13, 14, 15);
}

__device__ inline f32x8 wmma_bf16(bf16x16 a, bf16x16 b, f32x8 c) {
  // D = A(16x32 bf16) * B(32x16 bf16) + C(16x16 f32)
  return __builtin_amdgcn_wmma_f32_16x16x32_bf16(
      /*neg_a=*/false, a, /*neg_b=*/false, b,
      /*c_mod=*/(short)0, c, /*reuse_a=*/false, /*reuse_b=*/false);
}

// Per-lane 16B global -> LDS async copy (no VGPR round trip), tracked by ASYNCcnt.
__device__ inline void async_copy16(unsigned lds_off, const void* gptr) {
#if GEMM_USE_ASYNC
  asm volatile("global_load_async_to_lds_b128 %0, %1, off"
               :: "v"(lds_off), "v"(gptr) : "memory");
#else
  // Synchronous fallback: global load + ds store.
  bf16x8 v = *reinterpret_cast<const bf16x8*>(gptr);
  *reinterpret_cast<bf16x8*>((char*)nullptr + lds_off) = v;  // unused path
#endif
}

// ---------------------------------------------------------------- cast fp32->bf16
__global__ void k_cast_bf16(const float* __restrict__ s, __bf16* __restrict__ d, int n) {
  int i = blockIdx.x * blockDim.x + threadIdx.x;
  int stride = gridDim.x * blockDim.x;
  for (; i < n; i += stride) d[i] = f2bf(s[i]);
}

// ---------------------------------------------------------------- WMMA GEMM
// C[M][N] = A[M][K] * W[N][K]^T + bias[N]
// MODE 0: store bf16 row-major [M][N]
// MODE 1: store bf16 transposed [N][M]
// MODE 2: store f32 row-major [M][N]
//
// Block tile 64(M) x 256(N), K-step 32, double-buffered async LDS staging.
// 8 waves arranged 2(M) x 4(N); each wave computes 32x64 as 2x4 WMMA tiles.
// LDS rows padded to 40 elems (80B) so 16-lane b128 fragment reads are
// bank-conflict free (20*r mod 64 distinct for r=0..15).
template <int MODE>
__global__ __launch_bounds__(256)
void k_gemm_wmma(const __bf16* __restrict__ A, const __bf16* __restrict__ W,
                 const float* __restrict__ bias, void* __restrict__ outp,
                 int M, int N, int K) {
  constexpr int LDW = 40;                 // 32 + 8 pad
  __shared__ __bf16 As[2][64][LDW];       // 10 KB
  __shared__ __bf16 Ws[2][256][LDW];      // 40 KB

  const int tid  = threadIdx.x;
  const int lane = tid & 31;
  const int half = lane >> 4;
  const int l16  = lane & 15;
  const int wv   = tid >> 5;
  const int waveM = wv >> 2;              // 0..1
  const int waveN = wv & 3;               // 0..3

  const int blocksN = N >> 8;             // N / 256
  const int m0 = (blockIdx.x / blocksN) << 6;   // * 64
  const int n0 = (blockIdx.x % blocksN) << 8;   // * 256

  const int koffA = half * 16;            // A-frag: hi lanes hold K=16..31
  const int koffB = half * 16;            // B-frag: hi lanes hold K=16..31

  // Stage one 64x32 A tile + 256x32 W tile into LDS buffer `buf` (5 async ops/wave).
  auto load_tiles = [&](int buf, int k) {
    {
      const int row = tid >> 2, c = (tid & 3) * 8;
      async_copy16((unsigned)(size_t)&As[buf][row][c],
                   A + (size_t)(m0 + row) * K + k + c);
    }
#pragma unroll
    for (int c = 0; c < 4; ++c) {
      async_copy16((unsigned)(size_t)&Ws[buf][tid][c * 8],
                   W + (size_t)(n0 + tid) * K + k + c * 8);
    }
  };

  f32x8 acc[2][4];
#pragma unroll
  for (int i = 0; i < 2; ++i)
#pragma unroll
    for (int j = 0; j < 4; ++j) acc[i][j] = zero8();

  const int nk = K >> 5;                  // 32 iterations
  load_tiles(0, 0);
  load_tiles(1, 32);

  for (int it = 0; it < nk; ++it) {
    const int cur = it & 1;
    // Own async group for `cur` complete (<=5 keeps next buffer's loads in flight).
    if (it + 2 <= nk) asm volatile("s_wait_asynccnt 0x5" ::: "memory");
    else              asm volatile("s_wait_asynccnt 0x0" ::: "memory");
    __syncthreads();                      // all waves' tile data visible

    bf16x16 af[2];
#pragma unroll
    for (int i = 0; i < 2; ++i) {
      const __bf16* p = &As[cur][waveM * 32 + 16 * i + l16][koffA];
      af[i] = cat16(ld8(p), ld8(p + 8));
    }
    bf16x16 wf[4];
#pragma unroll
    for (int j = 0; j < 4; ++j) {
      const __bf16* p = &Ws[cur][waveN * 64 + 16 * j + l16][koffB];
      wf[j] = cat16(ld8(p), ld8(p + 8));
    }
#pragma unroll
    for (int i = 0; i < 2; ++i)
#pragma unroll
      for (int j = 0; j < 4; ++j)
        acc[i][j] = wmma_bf16(af[i], wf[j], acc[i][j]);

    __syncthreads();                      // everyone done reading `cur`
    if (it + 2 < nk) load_tiles(cur, (it + 2) * 32);
  }

#pragma unroll
  for (int j = 0; j < 4; ++j) {
    const int col = n0 + waveN * 64 + 16 * j + l16;
    const float bv = bias[col];
#pragma unroll
    for (int i = 0; i < 2; ++i) {
#pragma unroll
      for (int r = 0; r < 8; ++r) {
        const int row = m0 + waveM * 32 + 16 * i + r + 8 * half;
        const float v = acc[i][j][r] + bv;
        if (MODE == 0)
          ((__bf16*)outp)[(size_t)row * N + col] = f2bf(v);
        else if (MODE == 1)
          ((__bf16*)outp)[(size_t)col * M + row] = f2bf(v);
        else
          ((float*)outp)[(size_t)row * N + col] = v;
      }
    }
  }
}

// ---------------------------------------------------------------- flash attention
// Q, K: bf16 [B*T][H*64] row-major.  Vt: bf16 [H*64][B*T] (transposed).
// Out Ab: bf16 [B*T][H*64].  One wave handles 16 queries of one head.
__global__ __launch_bounds__(256)
void k_flash_attn(const __bf16* __restrict__ Qb, const __bf16* __restrict__ Kb,
                  const __bf16* __restrict__ Vt, __bf16* __restrict__ Ab) {
  __shared__ __bf16 Plds[8][16][32];   // per-wave 16x32 P tile (C-layout -> A-layout bounce)

  const int lane = threadIdx.x & 31;
  const int half = lane >> 4;
  const int l16  = lane & 15;
  const int wv   = threadIdx.x >> 5;
  const int gw   = blockIdx.x * 8 + wv;
  const int qtiles = Tc / 16;
  if (gw >= Bc * Hc * qtiles) return;
  const int qtile = gw % qtiles;
  const int bh    = gw / qtiles;
  const int b     = bh / Hc;
  const int h     = bh % Hc;
  const int q0    = qtile * 16;
  const int koffA = half * 16;
  const float scale = 0.125f;          // 1/sqrt(64)

  // Q A-fragments (d = 0..31 and 32..63)
  const __bf16* qrow = Qb + (size_t)(b * Tc + q0 + l16) * DMc + h * Dvc;
  const bf16x16 QA0 = cat16(ld8(qrow + koffA), ld8(qrow + koffA + 8));
  const bf16x16 QA1 = cat16(ld8(qrow + 32 + koffA), ld8(qrow + 32 + koffA + 8));

  f32x8 acc[4];
#pragma unroll
  for (int j = 0; j < 4; ++j) acc[j] = zero8();
  float row_m[8], row_l[8];
#pragma unroll
  for (int r = 0; r < 8; ++r) { row_m[r] = -1e30f; row_l[r] = 0.f; }

  const int nkb = qtile / 2 + 1;       // causal: keys < q0+16
  for (int kb = 0; kb < nkb; ++kb) {
    const int kbase = kb * 32;

    // ---- S = Q * K^T (two 16x16 score tiles for 32 keys)
    f32x8 S0 = zero8(), S1 = zero8();
    {
      const __bf16* k0 = Kb + (size_t)(b * Tc + kbase + l16) * DMc + h * Dvc + half * 16;
      const __bf16* k1 = Kb + (size_t)(b * Tc + kbase + 16 + l16) * DMc + h * Dvc + half * 16;
      S0 = wmma_bf16(QA0, cat16(ld8(k0), ld8(k0 + 8)), S0);
      S0 = wmma_bf16(QA1, cat16(ld8(k0 + 32), ld8(k0 + 40)), S0);
      S1 = wmma_bf16(QA0, cat16(ld8(k1), ld8(k1 + 8)), S1);
      S1 = wmma_bf16(QA1, cat16(ld8(k1 + 32), ld8(k1 + 40)), S1);
    }
    if (kb + 1 < nkb) {   // prefetch next key block (global_prefetch_b8)
      __builtin_prefetch(Kb + (size_t)(b * Tc + kbase + 32 + l16) * DMc + h * Dvc, 0, 1);
      __builtin_prefetch(Kb + (size_t)(b * Tc + kbase + 48 + l16) * DMc + h * Dvc, 0, 1);
    }

    // ---- online softmax (rows live across 16-lane halves in C layout)
    const int kcol0 = kbase + l16;
    const int kcol1 = kbase + 16 + l16;
#pragma unroll
    for (int r = 0; r < 8; ++r) {
      const int q = q0 + r + 8 * half;
      float s0 = S0[r] * scale; if (kcol0 > q) s0 = -1e30f;
      float s1 = S1[r] * scale; if (kcol1 > q) s1 = -1e30f;
      float mr = fmaxf(s0, s1);
      mr = fmaxf(mr, __shfl_xor(mr, 1, 32));
      mr = fmaxf(mr, __shfl_xor(mr, 2, 32));
      mr = fmaxf(mr, __shfl_xor(mr, 4, 32));
      mr = fmaxf(mr, __shfl_xor(mr, 8, 32));
      const float mn = fmaxf(row_m[r], mr);
      const float fs = __expf(row_m[r] - mn);
      row_m[r] = mn;
      const float p0 = __expf(s0 - mn);
      const float p1 = __expf(s1 - mn);
      float ls = p0 + p1;
      ls += __shfl_xor(ls, 1, 32);
      ls += __shfl_xor(ls, 2, 32);
      ls += __shfl_xor(ls, 4, 32);
      ls += __shfl_xor(ls, 8, 32);
      row_l[r] = row_l[r] * fs + ls;
      acc[0][r] *= fs; acc[1][r] *= fs; acc[2][r] *= fs; acc[3][r] *= fs;
      Plds[wv][r + 8 * half][l16]      = f2bf(p0);
      Plds[wv][r + 8 * half][16 + l16] = f2bf(p1);
    }

    // DS stores above feed cross-lane DS loads below: wait split DS counter.
    asm volatile("s_wait_dscnt 0" ::: "memory");

    const bf16x16 PA = cat16(*(const bf16x8*)&Plds[wv][l16][koffA],
                             *(const bf16x8*)&Plds[wv][l16][koffA + 8]);

    // ---- acc += P * V   (Vt transposed: lane's column d reads keys contiguously)
#pragma unroll
    for (int j = 0; j < 4; ++j) {
      const __bf16* vr = Vt + (size_t)(h * Dvc + j * 16 + l16) * (Bc * Tc)
                            + b * Tc + kbase + half * 16;
      acc[j] = wmma_bf16(PA, cat16(ld8(vr), ld8(vr + 8)), acc[j]);
    }
  }

  // ---- epilogue: divide by l, store bf16 [B*T][H*64]
#pragma unroll
  for (int r = 0; r < 8; ++r) {
    const float inv = 1.0f / row_l[r];
    const size_t rowoff = (size_t)(b * Tc + q0 + r + 8 * half) * DMc + h * Dvc;
#pragma unroll
    for (int j = 0; j < 4; ++j)
      Ab[rowoff + j * 16 + l16] = f2bf(acc[j][r] * inv);
  }
}

// ---------------------------------------------------------------- launch
extern "C" void kernel_launch(void* const* d_in, const int* in_sizes, int n_in,
                              void* d_out, int out_size, void* d_ws, size_t ws_size,
                              hipStream_t stream) {
  (void)in_sizes; (void)n_in; (void)out_size; (void)ws_size;
  const float* x  = (const float*)d_in[0];
  const float* Wq = (const float*)d_in[1];
  const float* bq = (const float*)d_in[2];
  const float* Wk = (const float*)d_in[3];
  const float* bk = (const float*)d_in[4];
  const float* Wv = (const float*)d_in[5];
  const float* bv = (const float*)d_in[6];
  const float* Wo = (const float*)d_in[7];
  const float* bo = (const float*)d_in[8];
  // d_in[9] = mask: causal structure is computed in-kernel.
  float* out = (float*)d_out;

  const int M = Bc * Tc;   // 8192
  const int N = DMc;       // 1024
  const int K = DMc;       // 1024

  char* ws = (char*)d_ws;
  size_t off = 0;
  auto alloc = [&](size_t bytes) {
    char* p = ws + off;
    off += (bytes + 255) & ~(size_t)255;
    return (void*)p;
  };
  __bf16* xb  = (__bf16*)alloc((size_t)M * K * 2);
  __bf16* wqb = (__bf16*)alloc((size_t)N * K * 2);
  __bf16* wkb = (__bf16*)alloc((size_t)N * K * 2);
  __bf16* wvb = (__bf16*)alloc((size_t)N * K * 2);
  __bf16* wob = (__bf16*)alloc((size_t)N * K * 2);
  __bf16* qb  = (__bf16*)alloc((size_t)M * N * 2);
  __bf16* kbf = (__bf16*)alloc((size_t)M * N * 2);
  __bf16* vtb = (__bf16*)alloc((size_t)M * N * 2);   // transposed [N][M]
  __bf16* ab  = (__bf16*)alloc((size_t)M * N * 2);

  // 1) casts
  k_cast_bf16<<<1024, 256, 0, stream>>>(x, xb, M * K);
  k_cast_bf16<<<512, 256, 0, stream>>>(Wq, wqb, N * K);
  k_cast_bf16<<<512, 256, 0, stream>>>(Wk, wkb, N * K);
  k_cast_bf16<<<512, 256, 0, stream>>>(Wv, wvb, N * K);
  k_cast_bf16<<<512, 256, 0, stream>>>(Wo, wob, N * K);

  // 2) projections: exact grid (no early-outs past barriers)
  const int gemmBlocks = (M / 64) * (N / 256);  // 512
  k_gemm_wmma<0><<<gemmBlocks, 256, 0, stream>>>(xb, wqb, bq, qb, M, N, K);
  k_gemm_wmma<0><<<gemmBlocks, 256, 0, stream>>>(xb, wkb, bk, kbf, M, N, K);
  k_gemm_wmma<1><<<gemmBlocks, 256, 0, stream>>>(xb, wvb, bv, vtb, M, N, K);

  // 3) fused causal attention
  k_flash_attn<<<(Bc * Hc * (Tc / 16)) / 8, 256, 0, stream>>>(qb, kbf, vtb, ab);

  // 4) output projection -> f32 d_out
  k_gemm_wmma<2><<<gemmBlocks, 256, 0, stream>>>(ab, wob, bo, out, M, N, K);
}
